// FCHiLo1_44839458570963
// MI455X (gfx1250) — matrix operand order repaired and checked by reading, not compiled, not run on hardware
//
#include <hip/hip_runtime.h>
#include <hip/hip_bf16.h>
#include <math.h>

// ---------------------------------------------------------------------------
// HiLo attention forward for MI455X (gfx1250), wave32 + WMMA f16->f32.
// Layouts: all activations token-major [B, T, C] (channel contiguous).
// ---------------------------------------------------------------------------

typedef __attribute__((ext_vector_type(16))) _Float16 v16h;
typedef __attribute__((ext_vector_type(8)))  float    v8f;

#define BDIM 8
#define HDIM 64
#define TOK  (HDIM * HDIM)       // 4096
#define CDIM 512
#define LOWH 32
#define LOWT (LOWH * LOWH)       // 1024
#define HEADS 4
#define HD    64
#define ATTN_SCALE 0.125f        // 64^-0.5

union FragH { v16h v; unsigned u[8]; };
union PackH { unsigned u; _Float16 h[2]; };

// ---- DPP16 row (16-lane) butterfly reductions: pure VALU, no LDS waits ----
template <int CTRL>
__device__ __forceinline__ float dpp_f(float x) {
  int r = __builtin_amdgcn_update_dpp(0, __builtin_bit_cast(int, x),
                                      CTRL, 0xF, 0xF, true);
  return __builtin_bit_cast(float, r);
}
__device__ __forceinline__ float row16_max(float v) {
  v = fmaxf(v, dpp_f<0xB1>(v));   // quad_perm [1,0,3,2]  (xor 1)
  v = fmaxf(v, dpp_f<0x4E>(v));   // quad_perm [2,3,0,1]  (xor 2)
  v = fmaxf(v, dpp_f<0x141>(v));  // row_half_mirror      (pairs 4<->4)
  v = fmaxf(v, dpp_f<0x140>(v));  // row_mirror           (pairs 8<->8)
  return v;
}
__device__ __forceinline__ float row16_sum(float v) {
  v += dpp_f<0xB1>(v);
  v += dpp_f<0x4E>(v);
  v += dpp_f<0x141>(v);
  v += dpp_f<0x140>(v);
  return v;
}

// ------------------------------- weight f32 -> f16 --------------------------
__global__ void __launch_bounds__(256)
cvt_kernel(const float* __restrict__ src, _Float16* __restrict__ dst, int n) {
  int i = blockIdx.x * blockDim.x + threadIdx.x;
  if (i < n) dst[i] = (_Float16)src[i];
}

// ------------------------------- pool + high --------------------------------
__global__ void __launch_bounds__(256)
prep_kernel(const float* __restrict__ x,
            float* __restrict__ low, float* __restrict__ high) {
  int idx = blockIdx.x * blockDim.x + threadIdx.x;   // BDIM*LOWT*CDIM threads
  int c  = idx & (CDIM - 1);
  int lt = (idx >> 9) & (LOWT - 1);
  int b  = idx >> 19;
  if (b >= BDIM) return;
  int ly = lt >> 5, lx = lt & 31;
  int t00 = (2 * ly) * HDIM + 2 * lx;
  size_t base = (size_t)b * TOK * CDIM;
  float a0 = x[base + (size_t)t00 * CDIM + c];
  float a1 = x[base + (size_t)(t00 + 1) * CDIM + c];
  float a2 = x[base + (size_t)(t00 + HDIM) * CDIM + c];
  float a3 = x[base + (size_t)(t00 + HDIM + 1) * CDIM + c];
  float avg = 0.25f * (a0 + a1 + a2 + a3);
  low[((size_t)b * LOWT + lt) * CDIM + c] = avg;
  high[base + (size_t)t00 * CDIM + c]              = avg - a0;
  high[base + (size_t)(t00 + 1) * CDIM + c]        = avg - a1;
  high[base + (size_t)(t00 + HDIM) * CDIM + c]     = avg - a2;
  high[base + (size_t)(t00 + HDIM + 1) * CDIM + c] = avg - a3;
}

// ------------------- fused depthwise 3x3 + pointwise GEMM -------------------
// Block: 128 threads (4 waves). Tile: 16 tokens (4x4 spatial) x all Cout.
template <int Cin, int Cout, int Hs, typename OutT>
__global__ void __launch_bounds__(128)
dsc_kernel(const float* __restrict__ in,         // [B, Hs*Hs, Cin]
           const float* __restrict__ dw,         // [Cin, 9]
           const float* __restrict__ dwb,        // [Cin]
           const _Float16* __restrict__ w16,     // [Cout, Cin]
           const float* __restrict__ pwb,        // [Cout]
           OutT* __restrict__ out, int outStride, int outOff) {
  constexpr int T = Hs * Hs;
  constexpr int PAD = 8;
  constexpr int LDA = Cin + PAD;
  constexpr int NG = Cout / 256;         // groups of (4 waves x 4 tiles x 16)
  __shared__ _Float16 ldsA[16 * LDA];

  const int tiles = T / 16;
  const int b    = blockIdx.x / tiles;
  const int tile = blockIdx.x % tiles;
  const int ty0 = (tile / (Hs / 4)) * 4;
  const int tx0 = (tile % (Hs / 4)) * 4;
  const int tid = threadIdx.x;

  // ---- depthwise stage ----
  for (int idx = tid; idx < 16 * Cin; idx += 128) {
    int t = idx / Cin;
    int c = idx % Cin;
    int gy = ty0 + (t >> 2), gx = tx0 + (t & 3);
    float s = dwb[c];
    #pragma unroll
    for (int dy = -1; dy <= 1; ++dy) {
      int y = gy + dy;
      if ((unsigned)y >= (unsigned)Hs) continue;
      #pragma unroll
      for (int dx = -1; dx <= 1; ++dx) {
        int xx = gx + dx;
        if ((unsigned)xx >= (unsigned)Hs) continue;
        s += in[((size_t)b * T + (size_t)y * Hs + xx) * Cin + c] *
             dw[c * 9 + (dy + 1) * 3 + (dx + 1)];
      }
    }
    ldsA[t * LDA + c] = (_Float16)s;
  }
  __syncthreads();

  // ---- GEMM stage ----
  const int wave = tid >> 5;
  const int lane = tid & 31;
  const int row  = lane & 15;
  const int col  = lane & 15;
  const int klo   = (lane < 16) ? 0 : 16;   // A-frag low-K per lane half
  const int bkoff = (lane < 16) ? 0 : 16;   // B-frag K offset per lane half

  for (int g = 0; g < NG; ++g) {
    v8f acc[4] = {{}, {}, {}, {}};
    const int nt0 = g * 16 + wave * 4;
    for (int kc = 0; kc < Cin; kc += 32) {
      FragH A;
      const _Float16* ap = &ldsA[row * LDA + kc + klo];
      #pragma unroll
      for (int i = 0; i < 4; ++i) A.u[i]     = *(const unsigned*)(ap + 2 * i);
      #pragma unroll
      for (int i = 0; i < 4; ++i) A.u[4 + i] = *(const unsigned*)(ap + 8 + 2 * i);
      #pragma unroll
      for (int j = 0; j < 4; ++j) {
        FragH B;
        const _Float16* bp = w16 + (size_t)((nt0 + j) * 16 + col) * Cin + kc + bkoff;
        __builtin_prefetch(bp + 32, 0, 1);
        #pragma unroll
        for (int i = 0; i < 8; ++i) B.u[i] = *(const unsigned*)(bp + 2 * i);
        acc[j] = __builtin_amdgcn_wmma_f32_16x16x32_f16(false, A.v, false, B.v,
                                                        (short)0, acc[j], false, false);
      }
    }
    #pragma unroll
    for (int j = 0; j < 4; ++j) {
      const float bias = pwb[(nt0 + j) * 16 + col];
      #pragma unroll
      for (int r = 0; r < 8; ++r) {
        int m = (lane < 16) ? r : (8 + r);
        int gy = ty0 + (m >> 2), gx = tx0 + (m & 3);
        size_t o = ((size_t)b * T + (size_t)gy * Hs + gx) * outStride
                 + outOff + (nt0 + j) * 16 + col;
        out[o] = (OutT)(acc[j][r] + bias);
      }
    }
  }
}

// ------------------------- low-frequency attention --------------------------
// Block: 4 waves = 4 consecutive 16-query tiles of the SAME (b, head). K/V are
// staged cooperatively into LDS once per 32-key chunk: K via async global->LDS
// DMA (ASYNCcnt), V transposed manually. Softmax row stats via DPP16.
__global__ void __launch_bounds__(128, 1)
low_attn_kernel(const _Float16* __restrict__ lq,   // [B,4096,256]
                const _Float16* __restrict__ lkv,  // [B,1024,512]
                float* __restrict__ outp) {        // [B,4096,256]
  constexpr int LDK = 72;   // K row pitch (halfs); 144 B = 16B-aligned rows
  constexpr int LDV = 40;   // Vt row pitch (halfs)
  constexpr int LDP = 40;   // P row pitch (halfs)
  __shared__ _Float16 ldsK[32 * LDK];
  __shared__ _Float16 ldsVt[64 * LDV];
  __shared__ _Float16 ldsP[4][16 * LDP];

  const int wave = threadIdx.x >> 5;
  const int lane = threadIdx.x & 31;
  const int tid  = threadIdx.x;

  const int qg   = blockIdx.x & 63;          // query group (4 tiles)
  const int head = (blockIdx.x >> 6) & 3;
  const int b    = blockIdx.x >> 8;
  const int qbase = (qg * 4 + wave) * 16;

  const int row   = lane & 15;
  const int col   = lane & 15;
  const int klo   = (lane < 16) ? 0 : 16;
  const int bkoff = (lane < 16) ? 0 : 16;

  // Q fragments (2 chunks of head-dim 32)
  FragH Aq[2];
  #pragma unroll
  for (int h = 0; h < 2; ++h) {
    const _Float16* qp = lq + ((size_t)b * TOK + qbase + row) * 256 + head * HD + h * 32 + klo;
    #pragma unroll
    for (int i = 0; i < 4; ++i) Aq[h].u[i]     = *(const unsigned*)(qp + 2 * i);
    #pragma unroll
    for (int i = 0; i < 4; ++i) Aq[h].u[4 + i] = *(const unsigned*)(qp + 8 + 2 * i);
  }

  v8f O[4] = {{}, {}, {}, {}};
  float rm[8], rl[8];
  #pragma unroll
  for (int r = 0; r < 8; ++r) { rm[r] = -INFINITY; rl[r] = 0.f; }

  _Float16* Pw = &ldsP[wave][0];
  const int skey = tid >> 2;          // staging: key 0..31
  const int sdg  = (tid & 3) * 16;    // staging: 16 dims per thread
  // LDS byte address (group-segment offset = low 32 bits of generic pointer)
  const unsigned kdst = (unsigned)(uintptr_t)&ldsK[skey * LDK + sdg];
  const int mrow = (lane < 16) ? 0 : 8;   // row base for this lane half

  for (int kb = 0; kb < LOWT; kb += 32) {
    // ---- cooperative K/V chunk staging ----
    {
      const _Float16* kg = lkv + ((size_t)b * LOWT + kb + skey) * 512 + head * HD + sdg;
      // K: async global->LDS DMA, 2 x b128 per thread (tracked by ASYNCcnt)
      asm volatile("global_load_async_to_lds_b128 %0, %1, off"
                   :: "v"(kdst), "v"(kg) : "memory");
      asm volatile("global_load_async_to_lds_b128 %0, %1, off"
                   :: "v"(kdst + 16u), "v"(kg + 8) : "memory");
      // V: manual transpose into LDS (two half-width groups: lower pressure)
      const _Float16* vg = kg + 256;
      #pragma unroll
      for (int gv = 0; gv < 2; ++gv) {
        unsigned vu[4];
        #pragma unroll
        for (int i = 0; i < 4; ++i) vu[i] = *(const unsigned*)(vg + 8 * gv + 2 * i);
        #pragma unroll
        for (int i = 0; i < 4; ++i) {
          PackH p; p.u = vu[i];
          int d = sdg + 8 * gv + 2 * i;
          ldsVt[d * LDV + skey]       = p.h[0];
          ldsVt[(d + 1) * LDV + skey] = p.h[1];
        }
      }
      asm volatile("s_wait_asynccnt 0" ::: "memory");
    }
    __syncthreads();

    // ---- S = Q K^T for two 16-key tiles ----
    v8f S0 = {}, S1 = {};
    #pragma unroll
    for (int h = 0; h < 2; ++h) {
      FragH B0, B1;
      const _Float16* k0 = &ldsK[col * LDK + h * 32 + bkoff];
      const _Float16* k1 = k0 + 16 * LDK;
      #pragma unroll
      for (int v = 0; v < 8; ++v) B0.u[v] = *(const unsigned*)(k0 + 2 * v);
      #pragma unroll
      for (int v = 0; v < 8; ++v) B1.u[v] = *(const unsigned*)(k1 + 2 * v);
      S0 = __builtin_amdgcn_wmma_f32_16x16x32_f16(false, Aq[h].v, false, B0.v, (short)0, S0, false, false);
      S1 = __builtin_amdgcn_wmma_f32_16x16x32_f16(false, Aq[h].v, false, B1.v, (short)0, S1, false, false);
    }

    // ---- fused per-row online softmax (DPP reductions, scalar temps) ----
    #pragma unroll
    for (int r = 0; r < 8; ++r) {
      float s0 = S0[r] * ATTN_SCALE, s1 = S1[r] * ATTN_SCALE;
      float cm = row16_max(fmaxf(s0, s1));
      float mn = fmaxf(rm[r], cm);
      float alpha = __expf(rm[r] - mn);
      rm[r] = mn;
      float p0 = __expf(s0 - mn);
      float p1 = __expf(s1 - mn);
      rl[r] = rl[r] * alpha + row16_sum(p0 + p1);
      O[0][r] *= alpha;
      O[1][r] *= alpha;
      O[2][r] *= alpha;
      O[3][r] *= alpha;
      Pw[(mrow + r) * LDP + col]      = (_Float16)p0;
      Pw[(mrow + r) * LDP + 16 + col] = (_Float16)p1;
    }

    // ---- repack P to A-fragment layout ----
    FragH AP;
    {
      const _Float16* ap = Pw + row * LDP + klo;
      #pragma unroll
      for (int i = 0; i < 4; ++i) AP.u[i]     = *(const unsigned*)(ap + 2 * i);
      #pragma unroll
      for (int i = 0; i < 4; ++i) AP.u[4 + i] = *(const unsigned*)(ap + 8 + 2 * i);
    }

    // ---- O += P V  (V transposed in LDS: contiguous K pairs) ----
    #pragma unroll
    for (int t = 0; t < 4; ++t) {
      FragH Bv;
      const _Float16* vb = &ldsVt[(t * 16 + col) * LDV + bkoff];
      #pragma unroll
      for (int v = 0; v < 8; ++v) Bv.u[v] = *(const unsigned*)(vb + 2 * v);
      O[t] = __builtin_amdgcn_wmma_f32_16x16x32_f16(false, AP.v, false, Bv.v, (short)0, O[t], false, false);
    }
    __syncthreads();   // protect ldsK/ldsVt for next chunk's staging
  }

  // ---- finalize & store ----
  #pragma unroll
  for (int t = 0; t < 4; ++t) {
    #pragma unroll
    for (int r = 0; r < 8; ++r) {
      outp[((size_t)b * TOK + qbase + mrow + r) * 256 + head * HD + t * 16 + col] =
          O[t][r] / rl[r];
    }
  }
}

// ------------------------- high-frequency attention -------------------------
__global__ void __launch_bounds__(256)
high_attn_kernel(const float* __restrict__ hqkv,  // [B,4096,768]
                 float* __restrict__ outh) {      // [B,4096,256]
  int idx = blockIdx.x * blockDim.x + threadIdx.x;  // 8*1024*4*4
  int q    = idx & 3;
  int head = (idx >> 2) & 3;
  int g    = (idx >> 4) & 1023;
  int b    = idx >> 14;
  if (b >= BDIM) return;
  int gy = g >> 5, gx = g & 31;
  int tok[4];
  #pragma unroll
  for (int j = 0; j < 4; ++j)
    tok[j] = (2 * gy + (j >> 1)) * HDIM + 2 * gx + (j & 1);

  const float* qp = hqkv + ((size_t)b * TOK + tok[q]) * 768 + head * HD;
  const float* kp[4];
  const float* vp[4];
  #pragma unroll
  for (int j = 0; j < 4; ++j) {
    kp[j] = hqkv + ((size_t)b * TOK + tok[j]) * 768 + 256 + head * HD;
    vp[j] = hqkv + ((size_t)b * TOK + tok[j]) * 768 + 512 + head * HD;
  }
  float s[4] = {0.f, 0.f, 0.f, 0.f};
  for (int d = 0; d < HD; ++d) {
    float qd = qp[d];
    #pragma unroll
    for (int j = 0; j < 4; ++j) s[j] += qd * kp[j][d];
  }
  float m = -INFINITY;
  #pragma unroll
  for (int j = 0; j < 4; ++j) { s[j] *= ATTN_SCALE; m = fmaxf(m, s[j]); }
  float p[4], sum = 0.f;
  #pragma unroll
  for (int j = 0; j < 4; ++j) { p[j] = __expf(s[j] - m); sum += p[j]; }
  float inv = 1.f / sum;
  #pragma unroll
  for (int j = 0; j < 4; ++j) p[j] *= inv;

  float* op = outh + ((size_t)b * TOK + tok[q]) * 256 + head * HD;
  for (int d = 0; d < HD; ++d) {
    float o = p[0] * vp[0][d] + p[1] * vp[1][d] + p[2] * vp[2][d] + p[3] * vp[3][d];
    op[d] = o;
  }
}

// ---------------------------------------------------------------------------
extern "C" void kernel_launch(void* const* d_in, const int* in_sizes, int n_in,
                              void* d_out, int out_size, void* d_ws, size_t ws_size,
                              hipStream_t stream) {
  (void)in_sizes; (void)n_in; (void)out_size; (void)ws_size;
  const float* x        = (const float*)d_in[0];
  const float* lq_dw    = (const float*)d_in[1];
  const float* lq_dwb   = (const float*)d_in[2];
  const float* lq_pw    = (const float*)d_in[3];
  const float* lq_pwb   = (const float*)d_in[4];
  const float* lkv_dw   = (const float*)d_in[5];
  const float* lkv_dwb  = (const float*)d_in[6];
  const float* lkv_pw   = (const float*)d_in[7];
  const float* lkv_pwb  = (const float*)d_in[8];
  const float* lp_dw    = (const float*)d_in[9];
  const float* lp_dwb   = (const float*)d_in[10];
  const float* lp_pw    = (const float*)d_in[11];
  const float* lp_pwb   = (const float*)d_in[12];
  const float* hq_dw    = (const float*)d_in[13];
  const float* hq_dwb   = (const float*)d_in[14];
  const float* hq_pw    = (const float*)d_in[15];
  const float* hq_pwb   = (const float*)d_in[16];
  const float* hp_dw    = (const float*)d_in[17];
  const float* hp_dwb   = (const float*)d_in[18];
  const float* hp_pw    = (const float*)d_in[19];
  const float* hp_pwb   = (const float*)d_in[20];
  float* outp = (float*)d_out;

  // ---- workspace layout ----
  char* ws = (char*)d_ws;
  size_t off = 0;
  auto alloc = [&](size_t bytes) { char* p = ws + off; off += (bytes + 255) & ~(size_t)255; return p; };
  float*    low   = (float*)   alloc((size_t)BDIM * LOWT * CDIM * 4);
  float*    high  = (float*)   alloc((size_t)BDIM * TOK * CDIM * 4);
  _Float16* lq16  = (_Float16*)alloc((size_t)BDIM * TOK * 256 * 2);
  _Float16* lkv16 = (_Float16*)alloc((size_t)BDIM * LOWT * 512 * 2);
  float*    hqkvf = (float*)   alloc((size_t)BDIM * TOK * 768 * 4);
  float*    latt  = (float*)   alloc((size_t)BDIM * TOK * 256 * 4);
  float*    hatt  = (float*)   alloc((size_t)BDIM * TOK * 256 * 4);
  _Float16* wlq   = (_Float16*)alloc((size_t)256 * 512 * 2);
  _Float16* wlkv  = (_Float16*)alloc((size_t)512 * 512 * 2);
  _Float16* whq   = (_Float16*)alloc((size_t)768 * 512 * 2);
  _Float16* wlp   = (_Float16*)alloc((size_t)256 * 256 * 2);
  _Float16* whp   = (_Float16*)alloc((size_t)256 * 256 * 2);

  // ---- weights -> f16 ----
  auto cvt = [&](const float* s, _Float16* d, int n) {
    cvt_kernel<<<(n + 255) / 256, 256, 0, stream>>>(s, d, n);
  };
  cvt(lq_pw, wlq, 256 * 512);
  cvt(lkv_pw, wlkv, 512 * 512);
  cvt(hq_pw, whq, 768 * 512);
  cvt(lp_pw, wlp, 256 * 256);
  cvt(hp_pw, whp, 256 * 256);

  // ---- pool / high ----
  {
    int n = BDIM * LOWT * CDIM;
    prep_kernel<<<n / 256, 256, 0, stream>>>(x, low, high);
  }

  // ---- q / kv / qkv projections (fused dw+pw, WMMA) ----
  dsc_kernel<512, 256, 64, _Float16><<<BDIM * (TOK / 16), 128, 0, stream>>>(
      x, lq_dw, lq_dwb, wlq, lq_pwb, lq16, 256, 0);
  dsc_kernel<512, 512, 32, _Float16><<<BDIM * (LOWT / 16), 128, 0, stream>>>(
      low, lkv_dw, lkv_dwb, wlkv, lkv_pwb, lkv16, 512, 0);
  dsc_kernel<512, 768, 64, float><<<BDIM * (TOK / 16), 128, 0, stream>>>(
      high, hq_dw, hq_dwb, whq, hq_pwb, hqkvf, 768, 0);

  // ---- attention ----
  low_attn_kernel<<<BDIM * HEADS * (TOK / 64), 128, 0, stream>>>(lq16, lkv16, latt);
  {
    int n = BDIM * LOWT * HEADS * 4;   // windows * heads * queries
    high_attn_kernel<<<n / 256, 256, 0, stream>>>(hqkvf, hatt);
  }

  // ---- output projections, fused concat into d_out [B,4096,512] ----
  dsc_kernel<256, 256, 64, float><<<BDIM * (TOK / 16), 128, 0, stream>>>(
      latt, lp_dw, lp_dwb, wlp, lp_pwb, outp, 512, 0);
  dsc_kernel<256, 256, 64, float><<<BDIM * (TOK / 16), 128, 0, stream>>>(
      hatt, hp_dw, hp_dwb, whp, hp_pwb, outp, 512, 256);
}
